// RegionProposalNetwork_8160437862425
// MI455X (gfx1250) — matrix operand order repaired and checked by reading, not compiled, not run on hardware
//
#include <hip/hip_runtime.h>
#include <hip/hip_bf16.h>
#include <float.h>

typedef __attribute__((ext_vector_type(16))) _Float16 v16h;
typedef __attribute__((ext_vector_type(8)))  float    v8f;
typedef __attribute__((ext_vector_type(4)))  unsigned u32x4;
typedef __attribute__((ext_vector_type(8)))  int      i32x8;
typedef __attribute__((ext_vector_type(4)))  int      i32x4;

#define C_IN      512
#define GH        50
#define GW        50
#define NPOS      2500
#define PADH      52
#define PADW      68
#define KANCH     9
#define NANCH     22500
#define NSORT     32768
#define PRE_TOPK  10000
#define POST_TOPK 2000
#define NMS_WORDS 320      /* ceil(10000/32)=313, padded */
#define IMG_SZ    800.0f
#define MIN_SIZE_F 16.0f
#define NMS_THR_F  0.7f

#define NCHUNKS   144      /* 9 taps x 16 ci-chunks of 32 */
#define BBUF_BYTES 16384   /* 16 N-tiles x 32 lanes x 16 halfs x 2B */

// ---------------------------------------------------------------------------
// 1) Pad (1-pixel halo) + fp32 -> f16 convert of the input feature map.
//    Layout: xpad[ci][r][c], r in [0,52), c in [0,68), zeros outside image.
// ---------------------------------------------------------------------------
__global__ void pad_convert_feat(const float* __restrict__ feat,
                                 _Float16* __restrict__ xpad) {
  int flat = blockIdx.x * 256 + threadIdx.x;        // 512*52*68 = 1,810,432 exact
  int ci = flat / (PADH * PADW);
  int rem = flat % (PADH * PADW);
  int r = rem / PADW, c = rem % PADW;
  float v = 0.0f;
  if (r >= 1 && r <= GH && c >= 1 && c <= GW)
    v = feat[ci * NPOS + (r - 1) * GW + (c - 1)];
  xpad[flat] = (_Float16)v;
}

// ---------------------------------------------------------------------------
// 2) Repack w_rpn (co,ci,3,3) fp32 into per-lane WMMA B-fragments (f16):
//    wfrag[(((tap*16 + cbi)*32 + nt)*32 + lane)*16 + e]
//    B 32x16 frag, this code's k order: lane&15 = N, K = e + 16*(lane>>4)
//    Each (tap, cbi, 16-ntile group) slab is 16KB CONTIGUOUS -> 1-D TDM tile.
// ---------------------------------------------------------------------------
__global__ void pack_weights(const float* __restrict__ w_rpn,
                             _Float16* __restrict__ wfrag) {
  int flat = blockIdx.x * 256 + threadIdx.x;        // 9<<18 = 2,359,296 exact
  int e    = flat & 15;
  int lane = (flat >> 4) & 31;
  int nt   = (flat >> 9) & 31;
  int cbi  = (flat >> 14) & 15;
  int tap  = flat >> 18;
  int k  = e + ((lane >> 4) << 4);
  int co = nt * 16 + (lane & 15);
  int ci = (cbi << 5) + k;
  wfrag[flat] = (_Float16)w_rpn[(co * C_IN + ci) * 9 + tap];
}

// ---------------------------------------------------------------------------
// Tensor Data Mover: 1-D contiguous copy of one 16KB B slab into LDS.
// D# group0: count=1, lds_addr, global_addr, type=2.
// D# group1: data_size=3 (8B), tensor_dim0=2048, tensor_dim1=1,
//            tile_dim0=2048, tile_dim1=1, dim0_stride=2048.
// ---------------------------------------------------------------------------
__device__ __forceinline__ void tdm_load_b_slab(unsigned lds_off,
                                                unsigned long long gaddr) {
  u32x4 g0 = (u32x4)0u;
  g0[0] = 1u;                                        // count=1 (valid D#)
  g0[1] = lds_off;                                   // lds_addr (bytes)
  g0[2] = (unsigned)(gaddr & 0xFFFFFFFFu);           // global_addr[31:0]
  g0[3] = (unsigned)((gaddr >> 32) & 0x01FFFFFFu) | (2u << 30);  // [56:32]|type=2
  i32x8 g1 = (i32x8)0;
  g1[0] = 0x30000;                                   // data_size = 3 (8 bytes)
  g1[1] = (int)(2048u << 16);                        // tensor_dim0 = 2048
  g1[2] = (int)(1u << 16);                           // tensor_dim1 = 1
  g1[3] = (int)(2048u << 16);                        // tile_dim0 = 2048
  g1[4] = 1;                                         // tile_dim1 = 1
  g1[5] = 2048;                                      // tensor_dim0_stride
  i32x4 gz = (i32x4)0;
#if defined(__clang_major__) && __clang_major__ >= 23
  i32x8 gz8 = (i32x8)0;
  __builtin_amdgcn_tensor_load_to_lds(g0, g1, gz, gz, gz8, 0);
#else
  __builtin_amdgcn_tensor_load_to_lds(g0, g1, gz, gz, 0);
#endif
}

// ---------------------------------------------------------------------------
// 3) 3x3 conv as implicit GEMM with v_wmma_f32_16x16x32_f16 + bias + ReLU.
//    Grid (200 M-tiles, 2 N-halves); block = 128 (4 waves); wave -> 4 N-tiles.
//    B operand streamed into LDS by the Tensor Data Mover (double-buffered,
//    TENSORcnt handshake); A staged in WMMA-fragment order via LDS.
// ---------------------------------------------------------------------------
__global__ __launch_bounds__(128)
void conv3x3_wmma(const _Float16* __restrict__ xpad,
                  const _Float16* __restrict__ wfrag,
                  const float* __restrict__ brpn,
                  float* __restrict__ rpn) {
  // Single shared symbol so the TDM lds_addr offsets are well-defined (base 0):
  // [0,1024)           : A fragment tile (32K x 16M, fragment order)
  // [1024, 1024+2*16K) : B slab double buffer
  __shared__ __align__(32) char lds_all[1024 + 2 * BBUF_BYTES];
  _Float16* ldsA = (_Float16*)lds_all;

  const int tile = blockIdx.x;                 // 0..199
  const int y    = tile >> 2;
  const int xb   = (tile & 3) << 4;
  const int tid  = threadIdx.x;
  const int wv   = tid >> 5;                   // 0..3
  const int lane = tid & 31;
  const int grpHi = lane >> 4;
  const int ntBase = blockIdx.y << 4;          // 16-ntile group
  const int ntl0 = wv << 2;                    // local ntile base (4 per wave)

  const unsigned long long wbase = (unsigned long long)(uintptr_t)wfrag;

  v8f acc[4] = {};

  // Prologue: DMA slab for chunk 0 into buffer 0.
  if (wv == 0)
    tdm_load_b_slab(1024u, wbase + (unsigned long long)(ntBase) * 1024ull);

  for (int tap = 0; tap < 9; ++tap) {
    const int ky = tap / 3, kx = tap % 3;
    const int yp = y + ky;
    for (int cbi = 0; cbi < 16; ++cbi) {
      const int idx = tap * 16 + cbi;
      const int cb  = cbi << 5;
      // Issue DMA for next slab into the other buffer; wait for current slab.
      if (wv == 0) {
        if (idx < NCHUNKS - 1) {
          unsigned lds_off = 1024u + (unsigned)((idx + 1) & 1) * BBUF_BYTES;
          unsigned long long goff =
              ((unsigned long long)(idx + 1) * 32ull + (unsigned)ntBase) * 1024ull;
          tdm_load_b_slab(lds_off, wbase + goff);
          __builtin_amdgcn_s_wait_tensorcnt(1);
        } else {
          __builtin_amdgcn_s_wait_tensorcnt(0);
        }
      }
      // Stage A tile (32 K x 16 M) into LDS in WMMA A-fragment order.
      // A 16x32 k order used here (matches B): lane<16 -> K 0-15 ; lane>=16 -> K 16-31
      #pragma unroll
      for (int u = 0; u < 4; ++u) {
        int t2 = tid + (u << 7);                 // 0..511
        int k  = t2 >> 4;
        int m  = t2 & 15;
        int grp = k >> 4;
        int e   = k & 15;
        ldsA[((m + (grp << 4)) << 4) + e] =
            xpad[((size_t)(cb + k) * PADH + yp) * PADW + (xb + kx + m)];
      }
      __syncthreads();
      v16h av = *(const v16h*)&ldsA[lane << 4];  // 2x ds_load_b128
      const _Float16* bb =
          (const _Float16*)(lds_all + 1024 + (idx & 1) * BBUF_BYTES);
      #pragma unroll
      for (int q = 0; q < 4; ++q) {
        v16h bv = *(const v16h*)&bb[(((ntl0 + q) << 5) + lane) << 4];
        acc[q] = __builtin_amdgcn_wmma_f32_16x16x32_f16(
            false, av, false, bv, (short)0, acc[q], false, false);
      }
      __syncthreads();
    }
  }
  // D layout: VGPR r -> M = r + 8*(lane>=16), N = lane&15
  const int ncol = lane & 15;
  #pragma unroll
  for (int q = 0; q < 4; ++q) {
    int co = (ntBase + ntl0 + q) * 16 + ncol;
    float bias = brpn[co];
    #pragma unroll
    for (int r = 0; r < 8; ++r) {
      int m = r + (grpHi << 3);
      int x = xb + m;
      if (x < GW)
        rpn[(size_t)co * NPOS + y * GW + x] = fmaxf(acc[q][r] + bias, 0.0f);
    }
  }
}

// ---------------------------------------------------------------------------
// 4) 1x1 convs (cls: 9ch, reg: 36ch), written in reference flat orders.
// ---------------------------------------------------------------------------
__global__ void conv1x1(const float* __restrict__ rpn,
                        const float* __restrict__ wcls, const float* __restrict__ bcls,
                        const float* __restrict__ wreg, const float* __restrict__ breg,
                        float* __restrict__ clsraw, float* __restrict__ regf) {
  int id = blockIdx.x * 256 + threadIdx.x;
  if (id >= 45 * NPOS) return;
  int oc = id / NPOS;
  int p  = id % NPOS;
  const float* w = (oc < 9) ? (wcls + oc * C_IN) : (wreg + (oc - 9) * C_IN);
  float s = 0.0f;
  #pragma unroll 4
  for (int ci = 0; ci < C_IN; ++ci) s += rpn[(size_t)ci * NPOS + p] * w[ci];
  if (oc < 9) {
    clsraw[p * 9 + oc] = s + bcls[oc];
  } else {
    int c = oc - 9;
    regf[(p * 9 + (c >> 2)) * 4 + (c & 3)] = s + breg[c];
  }
}

// ---------------------------------------------------------------------------
// 5) Anchor generation + box decode + sigmoid score.
// ---------------------------------------------------------------------------
__global__ void decode_anchors(const float* __restrict__ clsraw,
                               const float* __restrict__ regf,
                               float* __restrict__ props,
                               float* __restrict__ score) {
  int a = blockIdx.x * 256 + threadIdx.x;
  if (a >= NANCH) return;
  int p = a / 9, k = a % 9;
  int x = p % GW, y = p / GW;
  int ai = k / 3, si = k % 3;
  float ar = (ai == 0) ? 0.5f : (ai == 1 ? 1.0f : 2.0f);
  float sc = (si == 0) ? 128.0f : (si == 1 ? 256.0f : 512.0f);
  float hr = sqrtf(ar), wr = 1.0f / hr;
  float ws = wr * sc, hs = hr * sc;
  float a0 = (float)x + rintf(-ws * 0.5f);   // rintf == round-half-even (jnp.round)
  float a1 = (float)y + rintf(-hs * 0.5f);
  float a2 = (float)x + rintf( ws * 0.5f);
  float a3 = (float)y + rintf( hs * 0.5f);
  float w = a2 - a0, h = a3 - a1;
  float cx = a0 + 0.5f * w, cy = a1 + 0.5f * h;
  float dx = regf[a * 4 + 0], dy = regf[a * 4 + 1];
  float dw = regf[a * 4 + 2], dh = regf[a * 4 + 3];
  float pcx = dx * w + cx, pcy = dy * h + cy;
  float pw = expf(dw) * w, ph = expf(dh) * h;
  props[a * 4 + 0] = pcx - 0.5f * pw;
  props[a * 4 + 1] = pcy - 0.5f * ph;
  props[a * 4 + 2] = pcx + 0.5f * pw;
  props[a * 4 + 3] = pcy + 0.5f * ph;
  score[a] = 1.0f / (1.0f + expf(-clsraw[a]));
}

// ---------------------------------------------------------------------------
// 6) Bitonic sort of 32768 (score desc, idx asc) -> top-10000 prefix.
// ---------------------------------------------------------------------------
__global__ void sort_init(const float* __restrict__ score,
                          float* __restrict__ key, int* __restrict__ idx) {
  int i = blockIdx.x * 256 + threadIdx.x;
  if (i >= NSORT) return;
  key[i] = (i < NANCH) ? score[i] : -FLT_MAX;
  idx[i] = i;
}

__device__ __forceinline__ bool sort_before(float ka, int ia, float kb, int ib) {
  return (ka > kb) || (ka == kb && ia < ib);
}

__global__ void bitonic_step(float* __restrict__ key, int* __restrict__ idx,
                             int j, int k) {
  int i = blockIdx.x * 256 + threadIdx.x;
  if (i >= NSORT) return;
  int ixj = i ^ j;
  if (ixj <= i) return;
  float ki = key[i], kj = key[ixj];
  int   ii = idx[i], ij = idx[ixj];
  bool asc = ((i & k) == 0);   // "ascending" under the before-order comparator
  bool sw = asc ? sort_before(kj, ij, ki, ii) : sort_before(ki, ii, kj, ij);
  if (sw) { key[i] = kj; key[ixj] = ki; idx[i] = ij; idx[ixj] = ii; }
}

// ---------------------------------------------------------------------------
// 7) Gather top-10000, clip to image, size-filter.
// ---------------------------------------------------------------------------
__global__ void gather_clip(const float* __restrict__ key, const int* __restrict__ idx,
                            const float* __restrict__ props,
                            float* __restrict__ sbox, float* __restrict__ sscore,
                            int* __restrict__ valid) {
  int i = blockIdx.x * 256 + threadIdx.x;
  if (i >= PRE_TOPK) return;
  int id = idx[i];
  float x1 = fminf(fmaxf(props[id * 4 + 0], 0.0f), IMG_SZ);
  float y1 = fminf(fmaxf(props[id * 4 + 1], 0.0f), IMG_SZ);
  float x2 = fminf(fmaxf(props[id * 4 + 2], 0.0f), IMG_SZ);
  float y2 = fminf(fmaxf(props[id * 4 + 3], 0.0f), IMG_SZ);
  sbox[i * 4 + 0] = x1; sbox[i * 4 + 1] = y1;
  sbox[i * 4 + 2] = x2; sbox[i * 4 + 3] = y2;
  sscore[i] = key[i];
  valid[i] = (x2 - x1 >= MIN_SIZE_F) && (y2 - y1 >= MIN_SIZE_F);
}

// ---------------------------------------------------------------------------
// 8) NMS suppression-bit matrix (parallel over all pairs), then serial scan.
// ---------------------------------------------------------------------------
__global__ void nms_mask(const float* __restrict__ sbox, unsigned* __restrict__ mask) {
  int i = blockIdx.x;          // 0..9999
  int w = threadIdx.x;         // 0..319 (word index over j)
  float ix1 = sbox[i * 4 + 0], iy1 = sbox[i * 4 + 1];
  float ix2 = sbox[i * 4 + 2], iy2 = sbox[i * 4 + 3];
  float areaI = (ix2 - ix1) * (iy2 - iy1);
  unsigned bits = 0u;
  int j0 = w << 5;
  for (int jj = 0; jj < 32; ++jj) {
    int j = j0 + jj;
    if (j < PRE_TOPK && j > i) {
      float jx1 = sbox[j * 4 + 0], jy1 = sbox[j * 4 + 1];
      float jx2 = sbox[j * 4 + 2], jy2 = sbox[j * 4 + 3];
      float xx1 = fmaxf(ix1, jx1), yy1 = fmaxf(iy1, jy1);
      float xx2 = fminf(ix2, jx2), yy2 = fminf(iy2, jy2);
      float inter = fmaxf(xx2 - xx1, 0.0f) * fmaxf(yy2 - yy1, 0.0f);
      float areaJ = (jx2 - jx1) * (jy2 - jy1);
      float iou = inter / (areaI + areaJ - inter);
      if (iou > NMS_THR_F) bits |= (1u << jj);
    }
  }
  mask[(size_t)i * NMS_WORDS + w] = bits;
}

__global__ void nms_scan(const unsigned* __restrict__ mask,
                         const int* __restrict__ valid,
                         int* __restrict__ keep) {
  __shared__ unsigned removed[NMS_WORDS];
  __shared__ int s_kept;
  int t = threadIdx.x;         // block = 320 threads
  removed[t] = 0u;
  __syncthreads();
  for (int i = 0; i < PRE_TOPK; ++i) {
    if (t == 0) {
      int kk = valid[i] && !((removed[i >> 5] >> (i & 31)) & 1u);
      s_kept = kk;
      keep[i] = kk;
    }
    __syncthreads();
    if (s_kept) removed[t] |= mask[(size_t)i * NMS_WORDS + t];
    __syncthreads();
  }
}

// ---------------------------------------------------------------------------
// 9) Final "top-k 2000": kept entries are already score-sorted; sigmoid>0
//    always, so the reference reduces to stable compaction + zero padding.
// ---------------------------------------------------------------------------
__global__ void zero_out(float* __restrict__ out) {
  int i = blockIdx.x * 256 + threadIdx.x;
  if (i < POST_TOPK * 5) out[i] = 0.0f;    // 8000 box floats + 2000 scores
}

__global__ void compact(const int* __restrict__ keep,
                        const float* __restrict__ sbox,
                        const float* __restrict__ sscore,
                        float* __restrict__ out) {
  __shared__ int pref[1024];
  int t = threadIdx.x;
  int base = t * 10;
  int c = 0;
  for (int u = 0; u < 10; ++u) {
    int i = base + u;
    if (i < PRE_TOPK && keep[i]) c++;
  }
  pref[t] = c;
  __syncthreads();
  for (int off = 1; off < 1024; off <<= 1) {
    int add = (t >= off) ? pref[t - off] : 0;
    __syncthreads();
    pref[t] += add;
    __syncthreads();
  }
  int pos = pref[t] - c;       // exclusive prefix
  for (int u = 0; u < 10; ++u) {
    int i = base + u;
    if (i < PRE_TOPK && keep[i]) {
      if (pos < POST_TOPK) {
        out[pos * 4 + 0] = sbox[i * 4 + 0];
        out[pos * 4 + 1] = sbox[i * 4 + 1];
        out[pos * 4 + 2] = sbox[i * 4 + 2];
        out[pos * 4 + 3] = sbox[i * 4 + 3];
        out[4 * POST_TOPK + pos] = sscore[i];
      }
      pos++;
    }
  }
}

// ---------------------------------------------------------------------------
extern "C" void kernel_launch(void* const* d_in, const int* in_sizes, int n_in,
                              void* d_out, int out_size, void* d_ws, size_t ws_size,
                              hipStream_t stream) {
  const float* feat  = (const float*)d_in[1];
  const float* w_rpn = (const float*)d_in[3];
  const float* b_rpn = (const float*)d_in[4];
  const float* w_cls = (const float*)d_in[5];
  const float* b_cls = (const float*)d_in[6];
  const float* w_reg = (const float*)d_in[7];
  const float* b_reg = (const float*)d_in[8];
  float* out = (float*)d_out;

  size_t off = 0;
  auto alloc = [&](size_t bytes) -> void* {
    void* p = (char*)d_ws + off;
    off = (off + bytes + 255) & ~(size_t)255;
    return p;
  };
  _Float16* xpad   = (_Float16*)alloc((size_t)C_IN * PADH * PADW * 2);     // 3.45 MB
  _Float16* wfrag  = (_Float16*)alloc((size_t)9 * 16 * 32 * 32 * 16 * 2);  // 4.5 MB
  float*    rpn    = (float*)alloc((size_t)C_IN * NPOS * 4);               // 4.9 MB
  float*    clsraw = (float*)alloc((size_t)NANCH * 4);
  float*    regf   = (float*)alloc((size_t)NANCH * 4 * 4);
  float*    props  = (float*)alloc((size_t)NANCH * 4 * 4);
  float*    score  = (float*)alloc((size_t)NANCH * 4);
  float*    skey   = (float*)alloc((size_t)NSORT * 4);
  int*      sidx   = (int*)alloc((size_t)NSORT * 4);
  float*    sbox   = (float*)alloc((size_t)PRE_TOPK * 4 * 4);
  float*    sscore = (float*)alloc((size_t)PRE_TOPK * 4);
  int*      valid  = (int*)alloc((size_t)PRE_TOPK * 4);
  unsigned* mask   = (unsigned*)alloc((size_t)PRE_TOPK * NMS_WORDS * 4);   // 12.8 MB
  int*      keep   = (int*)alloc((size_t)PRE_TOPK * 4);
  (void)ws_size; (void)in_sizes; (void)n_in; (void)out_size;

  pad_convert_feat<<<7072, 256, 0, stream>>>(feat, xpad);
  pack_weights<<<9216, 256, 0, stream>>>(w_rpn, wfrag);
  conv3x3_wmma<<<dim3(200, 2), 128, 0, stream>>>(xpad, wfrag, b_rpn, rpn);
  conv1x1<<<440, 256, 0, stream>>>(rpn, w_cls, b_cls, w_reg, b_reg, clsraw, regf);
  decode_anchors<<<88, 256, 0, stream>>>(clsraw, regf, props, score);
  sort_init<<<128, 256, 0, stream>>>(score, skey, sidx);
  for (int k = 2; k <= NSORT; k <<= 1)
    for (int j = k >> 1; j > 0; j >>= 1)
      bitonic_step<<<128, 256, 0, stream>>>(skey, sidx, j, k);
  gather_clip<<<40, 256, 0, stream>>>(skey, sidx, props, sbox, sscore, valid);
  nms_mask<<<PRE_TOPK, NMS_WORDS, 0, stream>>>(sbox, mask);
  nms_scan<<<1, NMS_WORDS, 0, stream>>>(mask, valid, keep);
  zero_out<<<40, 256, 0, stream>>>(out);
  compact<<<1, 1024, 0, stream>>>(keep, sbox, sscore, out);
}
